// GATv2Conv_11433202942659
// MI455X (gfx1250) — matrix operand order repaired and checked by reading, not compiled, not run on hardware
//
#include <hip/hip_runtime.h>
#include <hip/hip_bf16.h>

// ---------------- problem constants (match reference) ----------------
#define N_NODES   50000
#define E_EDGES   1600000
#define IN_F      128
#define H_HEADS   4
#define F_OUT     32
#define HF        128            // H_HEADS * F_OUT
#define INV_CUTOFF 0.25f
#define ENV_A  (-36.0f)
#define ENV_B  ( 63.0f)
#define ENV_C  (-28.0f)

typedef __attribute__((ext_vector_type(16))) _Float16 v16h;
typedef __attribute__((ext_vector_type(8)))  float    v8f;

// ---------------- kernel 1: init outputs / softmax state ----------------
__global__ __launch_bounds__(256) void gat_init_kernel(float* __restrict__ rst,
                                                       float* __restrict__ node_max,
                                                       float* __restrict__ denom) {
  int i = blockIdx.x * 256 + threadIdx.x;
  if (i < N_NODES * HF) rst[i] = 0.0f;
  if (i < N_NODES * H_HEADS) {
    node_max[i] = -3.0e38f;
    denom[i]    = 0.0f;
  }
}

// ---------------- kernel 2: WMMA GEMM  feat = x @ W^T + b ----------------
// One wave (32 lanes) computes one 16x16 tile of [N_NODES, HF].
// blockIdx.x = M tile (3125), blockIdx.y = N tile (8), blockIdx.z = src/dst.
__global__ __launch_bounds__(32) void gat_feat_gemm_kernel(
    const float* __restrict__ x,
    const float* __restrict__ Wsrc, const float* __restrict__ bsrc,
    const float* __restrict__ Wdst, const float* __restrict__ bdst,
    float* __restrict__ feat_src, float* __restrict__ feat_dst) {
  const int mtile = blockIdx.x;
  const int ntile = blockIdx.y;
  const int which = blockIdx.z;

  const float* __restrict__ W    = which ? Wdst : Wsrc;
  const float* __restrict__ bias = which ? bdst : bsrc;
  float* __restrict__ out        = which ? feat_dst : feat_src;

  const int lane = threadIdx.x & 31;
  const int grp  = lane >> 4;     // which half-wave (0/1)
  const int l16  = lane & 15;

  const int m = mtile * 16 + l16; // A-matrix row owned by this lane
  const int n = ntile * 16 + l16; // B-matrix column owned by this lane

  v8f acc = {};
  #pragma unroll
  for (int k0 = 0; k0 < IN_F; k0 += 32) {
    v16h a, b;
    // A: 16x32 f16, lane=m, VGPR p holds K pair per ISA 7.12.2 layout
    #pragma unroll
    for (int p = 0; p < 8; ++p) {
      const int ka = k0 + (p >> 2) * 16 + grp * 8 + (p & 3) * 2;
      a[2 * p + 0] = (_Float16)x[m * IN_F + ka + 0];
      a[2 * p + 1] = (_Float16)x[m * IN_F + ka + 1];
      // B: 32x16 f16 = W^T tile; same K order as A (ISA 7.12.2 layout)
      const int kb = k0 + (p >> 2) * 16 + grp * 8 + (p & 3) * 2;
      b[2 * p + 0] = (_Float16)W[n * IN_F + kb + 0];
      b[2 * p + 1] = (_Float16)W[n * IN_F + kb + 1];
    }
    acc = __builtin_amdgcn_wmma_f32_16x16x32_f16(
        /*neg_a=*/false, a, /*neg_b=*/false, b,
        /*c_mod=*/(short)0, acc, /*reuse_a=*/false, /*reuse_b=*/false);
  }

  const float bn = bias[n];
  #pragma unroll
  for (int r = 0; r < 8; ++r) {
    const int row = mtile * 16 + r + grp * 8;  // C/D layout: VGPR r -> rows r, r+8
    out[(size_t)row * HF + ntile * 16 + l16] = acc[r] + bn;
  }
}

// ---------------- kernel 3: per-edge score + segment max ----------------
// One wave per edge; lane = feature f, loop over heads.
__global__ __launch_bounds__(256) void gat_edge_score_kernel(
    const float* __restrict__ feat_src, const float* __restrict__ feat_dst,
    const float* __restrict__ distance,
    const int* __restrict__ src, const int* __restrict__ dst,
    const float* __restrict__ attn, const float* __restrict__ alpha,
    const float* __restrict__ freq,
    float* __restrict__ score, float* __restrict__ node_max) {
  const int e = blockIdx.x * 8 + (threadIdx.x >> 5);
  if (e >= E_EDGES) return;
  const int lane = threadIdx.x & 31;

  const int s = src[e];
  const int d = dst[e];
  const float dd = distance[e] * INV_CUTOFF;
  const float d2 = dd * dd;
  const float d4 = d2 * d2;
  const float dp = d4 * d2 * dd;                       // d^7
  const float env = dd + ENV_A * dp + ENV_B * dp * dd + ENV_C * dp * d2;

  const float* __restrict__ fs = feat_src + (size_t)s * HF;
  const float* __restrict__ fd = feat_dst + (size_t)d * HF;

  #pragma unroll
  for (int h = 0; h < H_HEADS; ++h) {
    const float coeff = env * __sinf(freq[h] * dd);
    float ev = (fs[h * F_OUT + lane] + fd[h * F_OUT + lane]) * coeff;
    ev = (ev >= 0.0f) ? ev : alpha[h] * ev;            // PReLU per head
    float part = ev * attn[h * F_OUT + lane];
    #pragma unroll
    for (int off = 16; off > 0; off >>= 1)
      part += __shfl_xor(part, off, 32);
    if (lane == 0) {
      score[(size_t)e * H_HEADS + h] = part;
      atomicMax(&node_max[d * H_HEADS + h], part);     // L2-resident fp atomic max
    }
  }
}

// ---------------- kernel 4: exp(score - max) + segment sum ----------------
__global__ __launch_bounds__(256) void gat_edge_exp_kernel(
    const int* __restrict__ dst, const float* __restrict__ node_max,
    float* __restrict__ score, float* __restrict__ denom) {
  const int i = blockIdx.x * 256 + threadIdx.x;
  if (i >= E_EDGES * H_HEADS) return;
  const int e = i >> 2;
  const int h = i & 3;
  const int d = dst[e];
  const float ex = __expf(score[i] - node_max[d * H_HEADS + h]);
  score[i] = ex;                                       // in-place: score -> ex
  atomicAdd(&denom[d * H_HEADS + h], ex);
}

// ---------------- kernel 5: normalize + scatter-sum aggregation ----------------
__global__ __launch_bounds__(256) void gat_edge_aggregate_kernel(
    const float* __restrict__ feat_src,
    const int* __restrict__ src, const int* __restrict__ dst,
    const float* __restrict__ score, const float* __restrict__ denom,
    float* __restrict__ rst) {
  const int e = blockIdx.x * 8 + (threadIdx.x >> 5);
  if (e >= E_EDGES) return;
  const int lane = threadIdx.x & 31;

  const int s = src[e];
  const int d = dst[e];
  const float* __restrict__ fs = feat_src + (size_t)s * HF;
  float* __restrict__ rd = rst + (size_t)d * HF;

  #pragma unroll
  for (int h = 0; h < H_HEADS; ++h) {
    const float a = score[(size_t)e * H_HEADS + h] / denom[d * H_HEADS + h];
    atomicAdd(&rd[h * F_OUT + lane], fs[h * F_OUT + lane] * a);
  }
}

// ---------------- launcher ----------------
extern "C" void kernel_launch(void* const* d_in, const int* in_sizes, int n_in,
                              void* d_out, int out_size, void* d_ws, size_t ws_size,
                              hipStream_t stream) {
  const float* x        = (const float*)d_in[0];
  const float* distance = (const float*)d_in[1];
  const float* W_src    = (const float*)d_in[2];
  const float* b_src    = (const float*)d_in[3];
  const float* W_dst    = (const float*)d_in[4];
  const float* b_dst    = (const float*)d_in[5];
  const float* attn     = (const float*)d_in[6];
  const float* alpha    = (const float*)d_in[7];
  const float* freq     = (const float*)d_in[8];
  const int*   src      = (const int*)d_in[9];
  const int*   dst      = (const int*)d_in[10];
  float* rst            = (float*)d_out;

  // workspace layout (bytes), 256-aligned sections
  char* ws = (char*)d_ws;
  float* feat_src = (float*)(ws + 0);            // 25,600,000 B
  float* feat_dst = (float*)(ws + 25600000);     // 25,600,000 B
  float* score    = (float*)(ws + 51200000);     // 25,600,000 B  [E,H]
  float* node_max = (float*)(ws + 76800000);     //    800,000 B  [N,H]
  float* denom    = (float*)(ws + 77600256);     //    800,000 B  [N,H]

  // 1) init rst / node_max / denom
  gat_init_kernel<<<(N_NODES * HF + 255) / 256, 256, 0, stream>>>(rst, node_max, denom);

  // 2) WMMA feature projections (both src & dst via blockIdx.z)
  gat_feat_gemm_kernel<<<dim3(N_NODES / 16, HF / 16, 2), 32, 0, stream>>>(
      x, W_src, b_src, W_dst, b_dst, feat_src, feat_dst);

  // 3) edge scores + per-(dst,head) max
  gat_edge_score_kernel<<<E_EDGES / 8, 256, 0, stream>>>(
      feat_src, feat_dst, distance, src, dst, attn, alpha, freq, score, node_max);

  // 4) exp + denom
  gat_edge_exp_kernel<<<(E_EDGES * H_HEADS) / 256, 256, 0, stream>>>(
      dst, node_max, score, denom);

  // 5) normalize + aggregate
  gat_edge_aggregate_kernel<<<E_EDGES / 8, 256, 0, stream>>>(
      feat_src, src, dst, score, denom, rst);
}
